// NeuralNetwork_86990267613505
// MI455X (gfx1250) — matrix operand order repaired and checked by reading, not compiled, not run on hardware
//
#include <hip/hip_runtime.h>
#include <hip/hip_bf16.h>
#include <stdint.h>

// ---------------------------------------------------------------------------
// CDNA5 (gfx1250) implementation.
// Roofline: ~27 GFLOP over ~230 MB unique bytes (AI ~120 FLOP/B) -> needs
// ~2.8 PFLOP/s matrix throughput to stay HBM-bound. Only the bf16 WMMA path
// (v_wmma_f32_16x16x32_bf16, fp32 accum) clears that; fp32 WMMA (K=4) would
// be ~8x too slow. Data stays fp32 in memory (no extra bytes), converted to
// bf16 with a single v_cvt_pk_bf16_f32 per pair (inline asm: the builtin name
// is absent on this toolchain but the instruction exists) during LDS staging.
// The GEMMs are skinny (M*N=256K) with fat K -> split-K across blockIdx.z
// (deterministic partials + fixed-order reduction, no FP atomics) so enough
// workgroups exist to saturate 23.3 TB/s.
// ---------------------------------------------------------------------------

typedef __attribute__((ext_vector_type(16))) __bf16 v16bf;
typedef __attribute__((ext_vector_type(2)))  __bf16 v2bf;
typedef __attribute__((ext_vector_type(8)))  float  v8f;

union FragAB {
    uint4 q[2];
    v16bf v;
};

// fp32 pair -> packed bf16 dword (RNE) in ONE VALU op.
__device__ __forceinline__ uint32_t pack_bf16(float lo, float hi) {
#if __has_builtin(__builtin_amdgcn_cvt_pk_bf16_f32)
    union { v2bf v; uint32_t u; } cv;
    cv.v = __builtin_amdgcn_cvt_pk_bf16_f32(lo, hi);
    return cv.u;
#else
    uint32_t u;
    asm("v_cvt_pk_bf16_f32 %0, %1, %2" : "=v"(u) : "v"(lo), "v"(hi));
    return u;
#endif
}

__device__ __forceinline__ uint4 cvt8(const float4 f0, const float4 f1) {
    uint4 s;
    s.x = pack_bf16(f0.x, f0.y);
    s.y = pack_bf16(f0.z, f0.w);
    s.z = pack_bf16(f1.x, f1.y);
    s.w = pack_bf16(f1.z, f1.w);
    return s;
}

// Partial GEMM: Cp[z] = A[256,K][:, kr] @ W[N,K][:, kr]^T for this block's
// K-range kr. Block tile 128(M) x 128(N); 8 waves each own 64x32 (4x2 WMMA
// tiles); K stepped by 64 (two WMMA slabs per LDS stage). Global fp32 tiles
// are double-buffered in registers so HBM latency overlaps the WMMAs.
__global__ __launch_bounds__(256)
void gemm_bf16_wmma(const float* __restrict__ A, const float* __restrict__ W,
                    float* __restrict__ Cp, int N, int K, int kc) {
    __shared__ uint32_t Als[128 * 32];   // 16 KB: A tile 128x64 bf16
    __shared__ uint32_t Bls[128 * 32];   // 16 KB: W tile 128x64 bf16

    const int tid  = threadIdx.x;
    const int lane = tid & 31;
    const int wave = tid >> 5;                 // 0..7 -> 2x4 wave grid
    const int wrow = (wave >> 2) * 64;
    const int wcol = (wave & 3) * 32;
    const int mBase = blockIdx.y * 128;
    const int nBase = blockIdx.x * 128;
    const int sel = lane >> 4;
    const int l15 = lane & 15;

    const int kStart = blockIdx.z * kc;
    const int kEnd   = kStart + kc;

    // staging role: 2 threads per tile row, 32 contiguous cols each
    const int r = tid >> 1;
    const int h = (tid & 1) << 5;
    const float* pa = A + (size_t)(mBase + r) * (size_t)K + (size_t)(kStart + h);
    const float* pb = W + (size_t)(nBase + r) * (size_t)K + (size_t)(kStart + h);
    uint32_t* qa = Als + r * 32 + (h >> 1);
    uint32_t* qb = Bls + r * 32 + (h >> 1);

    float4 apf[8], bpf[8];
#pragma unroll
    for (int i = 0; i < 8; ++i) {
        apf[i] = ((const float4*)pa)[i];
        bpf[i] = ((const float4*)pb)[i];
    }

    v8f acc[4][2] = {};

    for (int k0 = kStart; k0 < kEnd; k0 += 64) {
        __syncthreads();                       // LDS free of prior readers
#pragma unroll
        for (int i = 0; i < 4; ++i) {
            ((uint4*)qa)[i] = cvt8(apf[2 * i], apf[2 * i + 1]);
            ((uint4*)qb)[i] = cvt8(bpf[2 * i], bpf[2 * i + 1]);
        }
        __syncthreads();

        if (k0 + 64 < kEnd) {                  // register prefetch of next tile
            pa += 64; pb += 64;
            __builtin_prefetch(pa + 64, 0, 1); // L2 prefetch two tiles ahead
            __builtin_prefetch(pb + 64, 0, 1);
#pragma unroll
            for (int i = 0; i < 8; ++i) {
                apf[i] = ((const float4*)pa)[i];
                bpf[i] = ((const float4*)pb)[i];
            }
        }

#pragma unroll
        for (int s = 0; s < 2; ++s) {          // two K=32 slabs per stage
            // B frag: lane n<16 -> K 0-7 & 16-23; lane n+16 -> K 8-15 & 24-31
            FragAB bf[2];
#pragma unroll
            for (int nt = 0; nt < 2; ++nt) {
                const uint32_t* p = Bls + (wcol + nt * 16 + l15) * 32 + s * 16;
                bf[nt].q[0] = *(const uint4*)(p + sel * 4);
                bf[nt].q[1] = *(const uint4*)(p + 8 + sel * 4);
            }
            // A frag: lane m<16 -> K 0-7 (v0-3) & 16-23 (v4-7); +16 -> 8-15 & 24-31
#pragma unroll
            for (int mt = 0; mt < 4; ++mt) {
                const uint32_t* p = Als + (wrow + mt * 16 + l15) * 32 + s * 16;
                FragAB af;
                af.q[0] = *(const uint4*)(p + sel * 4);
                af.q[1] = *(const uint4*)(p + 8 + sel * 4);
#pragma unroll
                for (int nt = 0; nt < 2; ++nt) {
                    acc[mt][nt] = __builtin_amdgcn_wmma_f32_16x16x32_bf16(
                        false, af.v, false, bf[nt].v, (short)0, acc[mt][nt],
                        false, false);
                }
            }
        }
    }

    // Raw partial store. C/D layout: VGPR r, lane L -> M = r + 8*(L>>4), N = L&15.
    float* Cz = Cp + (size_t)blockIdx.z * 256 * (size_t)N;
#pragma unroll
    for (int mt = 0; mt < 4; ++mt) {
        const int rbase = mBase + wrow + mt * 16 + (sel << 3);
#pragma unroll
        for (int nt = 0; nt < 2; ++nt) {
            const int col = nBase + wcol + nt * 16 + l15;
#pragma unroll
            for (int rr = 0; rr < 8; ++rr)
                Cz[(size_t)(rbase + rr) * (size_t)N + col] = acc[mt][nt][rr];
        }
    }
}

// Deterministic split-K reduction + bias + optional tanh. Fixed summation
// order -> bitwise reproducible. ACT: 0 identity, 1 tanh.
template <int ACT>
__global__ __launch_bounds__(256)
void reduce_bias_act(const float* __restrict__ Cp, const float* __restrict__ bias,
                     float* __restrict__ C, int N, int zc) {
    const int idx = blockIdx.x * 256 + threadIdx.x;    // over 256*N elements
    float s = 0.0f;
    for (int z = 0; z < zc; ++z)
        s += Cp[(size_t)z * 256 * (size_t)N + idx];
    const int col = idx & (N - 1);                     // N is a power of two
    if (bias) s += bias[col];
    if (ACT == 1) s = tanhf(s);
    C[idx] = s;
}

// cx[b] = sigmoid( dot(cxh[b, :512], Wc2[:512]) ); one block (64 thr) per row
__global__ __launch_bounds__(64)
void ctx_kernel(const float* __restrict__ cxh, const float* __restrict__ Wc2,
                float* __restrict__ cx) {
    const int b = blockIdx.x;
    const int t = threadIdx.x;
    float s = 0.0f;
    for (int i = t; i < 512; i += 64)
        s += cxh[(size_t)b * 512 + i] * Wc2[i];
    __shared__ float red[2];
#pragma unroll
    for (int off = 16; off > 0; off >>= 1)
        s += __shfl_down(s, off, 32);                  // wave32 reduction
    if ((t & 31) == 0) red[t >> 5] = s;
    __syncthreads();
    if (t == 0) {
        const float tot = red[0] + red[1];
        cx[b] = 1.0f / (1.0f + __expf(-tot));
    }
}

// Dynamic-k WTA, in place (row snapshotted to LDS first). Stable tie-break
// (j < i) matches jnp.argsort rank semantics. Inner loop reads ds_load_b128.
__global__ __launch_bounds__(256)
void kwta_kernel(const float* __restrict__ x, const float* __restrict__ cx,
                 float* __restrict__ y, int n) {
    __shared__ float row[1024];
    const int b = blockIdx.x;
    const int t = threadIdx.x;
    for (int i = t; i < n; i += 256) row[i] = x[(size_t)b * n + i];
    __syncthreads();
    const int k = (int)(cx[b] * (float)n);             // torch .int() truncation
    for (int i = t; i < n; i += 256) {
        const float v = row[i];
        int rank = 0;
        for (int j = 0; j < n; j += 4) {
            const float4 u = *(const float4*)&row[j];
            rank += (int)(u.x > v) | ((int)(u.x == v) & (int)(j     < i));
            rank += (int)(u.y > v) | ((int)(u.y == v) & (int)(j + 1 < i));
            rank += (int)(u.z > v) | ((int)(u.z == v) & (int)(j + 2 < i));
            rank += (int)(u.w > v) | ((int)(u.w == v) & (int)(j + 3 < i));
        }
        y[(size_t)b * n + i] = (rank < k) ? v : 0.0f;
    }
}

extern "C" void kernel_launch(void* const* d_in, const int* in_sizes, int n_in,
                              void* d_out, int out_size, void* d_ws, size_t ws_size,
                              hipStream_t stream) {
    (void)in_sizes; (void)n_in; (void)out_size;

    const float* input = (const float*)d_in[0];   // [256, 32768]
    const float* W_c1  = (const float*)d_in[1];   // [512, 32768]
    const float* b_c1  = (const float*)d_in[2];   // [512]
    const float* W_c2  = (const float*)d_in[3];   // [1, 512]
    const float* W1    = (const float*)d_in[4];   // [1024, 32768]
    const float* b1    = (const float*)d_in[5];   // [1024]
    const float* W2    = (const float*)d_in[6];   // [512, 1024]
    const float* b2    = (const float*)d_in[7];   // [512]
    const float* W3    = (const float*)d_in[8];   // [1024, 512]
    const float* b3    = (const float*)d_in[9];   // [1024]
    const float* W4    = (const float*)d_in[10];  // [1024, 1024]
    float* out = (float*)d_out;                   // [256, 1024]

    // fp32 workspace layout
    char* base = (char*)d_ws;
    float* cxh = (float*)(base);                               // 256*512
    float* cx  = (float*)(base + 524288);                      // 256
    float* x1  = (float*)(base + 525312);                      // 256*1024
    float* x2  = (float*)(base + 1573888);                     // 256*512
    float* x3  = (float*)(base + 2098176);                     // 256*1024
    const size_t partOff = 3146752;
    float* part = (float*)(base + partOff);
    const size_t availPart = (ws_size > partOff) ? (ws_size - partOff) : 0;

    // Deterministic: zc depends only on ws_size (fixed per process).
    auto launch_gemm = [&](const float* Ain, const float* Wm, const float* bias,
                           float* Cout, int N, int K, int zcWant, int act) {
        const size_t oneSlab = (size_t)256 * (size_t)N * 4u;
        int maxz = (availPart >= oneSlab) ? (int)(availPart / oneSlab) : 0;
        int zc = zcWant;
        while (zc > 1 && (zc > maxz || (K / 64) % zc != 0)) zc >>= 1;
        float* Cp = (maxz >= 1) ? part : Cout;   // fallback: in-place, zc==1
        if (maxz < 1) zc = 1;
        const int kc = K / zc;
        gemm_bf16_wmma<<<dim3(N / 128, 2, zc), dim3(256), 0, stream>>>(
            Ain, Wm, Cp, N, K, kc);
        if (act)
            reduce_bias_act<1><<<dim3(N), dim3(256), 0, stream>>>(Cp, bias, Cout, N, zc);
        else
            reduce_bias_act<0><<<dim3(N), dim3(256), 0, stream>>>(Cp, bias, Cout, N, zc);
    };

    // context gate: cxh = tanh(input @ W_c1^T + b_c1)        [256, 512]
    launch_gemm(input, W_c1, b_c1, cxh, 512, 32768, 8, 1);
    // cx = sigmoid(cxh @ W_c2^T)                              [256]
    ctx_kernel<<<dim3(256), dim3(64), 0, stream>>>(cxh, W_c2, cx);

    // x1 = input @ W1^T + b1                                  [256, 1024]
    launch_gemm(input, W1, b1, x1, 1024, 32768, 8, 0);
    kwta_kernel<<<dim3(256), dim3(256), 0, stream>>>(x1, cx, x1, 1024);

    // x2 = x1 @ W2^T + b2                                     [256, 512]
    launch_gemm(x1, W2, b2, x2, 512, 1024, 4, 0);
    kwta_kernel<<<dim3(256), dim3(256), 0, stream>>>(x2, cx, x2, 512);

    // x3 = x2 @ W3^T + b3                                     [256, 1024]
    launch_gemm(x2, W3, b3, x3, 1024, 512, 2, 0);
    kwta_kernel<<<dim3(256), dim3(256), 0, stream>>>(x3, cx, x3, 1024);

    // out = x3 @ W4^T                                         [256, 1024]
    launch_gemm(x3, W4, nullptr, out, 1024, 1024, 4, 0);
}
